// TimeSeriesConvolutionalGraphModel_20667382628588
// MI455X (gfx1250) — compile-verified, matrix operand order checked by reading, not run on hardware
//
#include <hip/hip_runtime.h>

#define N_NODES   50000
#define N_EDGES   800000
#define DF        128
#define NGRAPHS   50
#define NLAYERS   3

typedef __attribute__((ext_vector_type(16))) _Float16 v16h;
typedef __attribute__((ext_vector_type(8)))  _Float16 v8h;
typedef __attribute__((ext_vector_type(8)))  float    v8f;

// ---------------------------------------------------------------- utilities
__global__ void fill_f32(float* __restrict__ p, float v, long n) {
    long i = (long)blockIdx.x * blockDim.x + threadIdx.x;
    long stride = (long)gridDim.x * blockDim.x;
    for (; i < n; i += stride) p[i] = v;
}

__global__ void degree_accum(const int* __restrict__ dst, float* __restrict__ deg, int nE) {
    int i = blockIdx.x * blockDim.x + threadIdx.x;
    if (i < nE) atomicAdd(&deg[dst[i]], 1.0f);
}

__global__ void rsqrt_k(float* __restrict__ deg, int n) {
    int i = blockIdx.x * blockDim.x + threadIdx.x;
    if (i < n) deg[i] = rsqrtf(deg[i]);   // deg >= 1 always (self-loops)
}

// ---------------------------------------------------------------- WMMA GEMM
// H[N,128] = X[N,128] @ W[128,128]; one wave computes a 16x128 output tile.
__global__ void __launch_bounds__(256)
gcn_gemm_wmma(const float* __restrict__ X, const float* __restrict__ W,
              float* __restrict__ H, int nRows) {
    // W converted to f16, stored transposed: sWt[n*128 + k] = W[k*128 + n]
    __shared__ _Float16 sWt[DF * DF];     // 32 KB
    for (int i = threadIdx.x; i < DF * DF; i += 256) {
        int k = i >> 7, n = i & 127;
        sWt[n * DF + k] = (_Float16)W[i];
    }
    __syncthreads();

    int wave = threadIdx.x >> 5;
    int lane = threadIdx.x & 31;
    int row0 = (blockIdx.x * 8 + wave) * 16;
    if (row0 >= nRows) return;            // wave-uniform; EXEC stays full below

    int mrow  = lane & 15;
    int khalf = (lane < 16) ? 0 : 8;      // per-lane K half per A/B fragment layout
    const float* xrow = X + (size_t)(row0 + mrow) * DF;

    v8f c[8];
#pragma unroll
    for (int t = 0; t < 8; ++t) c[t] = (v8f){0.f,0.f,0.f,0.f,0.f,0.f,0.f,0.f};

#pragma unroll
    for (int kt = 0; kt < 4; ++kt) {
        int k0 = kt * 32;
        // ---- A fragment: 16x32 f16 (lane<16: K=k0+[0..7],k0+16+[0..7]; lane>=16: +8)
        float4 a0 = ((const float4*)(xrow + k0 + khalf))[0];
        float4 a1 = ((const float4*)(xrow + k0 + khalf))[1];
        float4 a2 = ((const float4*)(xrow + k0 + 16 + khalf))[0];
        float4 a3 = ((const float4*)(xrow + k0 + 16 + khalf))[1];
        v16h a;
        a[0]=(_Float16)a0.x; a[1]=(_Float16)a0.y; a[2]=(_Float16)a0.z; a[3]=(_Float16)a0.w;
        a[4]=(_Float16)a1.x; a[5]=(_Float16)a1.y; a[6]=(_Float16)a1.z; a[7]=(_Float16)a1.w;
        a[8]=(_Float16)a2.x; a[9]=(_Float16)a2.y; a[10]=(_Float16)a2.z; a[11]=(_Float16)a2.w;
        a[12]=(_Float16)a3.x; a[13]=(_Float16)a3.y; a[14]=(_Float16)a3.z; a[15]=(_Float16)a3.w;

#pragma unroll
        for (int nt = 0; nt < 8; ++nt) {
            int n = nt * 16 + (lane & 15);
            // ---- B fragment: contiguous f16 runs in transposed LDS -> ds_load_b128
            v8h blo = *(const v8h*)&sWt[n * DF + k0 + khalf];
            v8h bhi = *(const v8h*)&sWt[n * DF + k0 + 16 + khalf];
            v16h b;
#pragma unroll
            for (int j = 0; j < 8; ++j) { b[j] = blo[j]; b[8 + j] = bhi[j]; }
            c[nt] = __builtin_amdgcn_wmma_f32_16x16x32_f16(
                        false, a, false, b, (short)0, c[nt], false, false);
        }
    }

    // ---- store D: VGPR r -> M=r (lanes 0-15), M=r+8 (lanes 16-31); N = lane&15
#pragma unroll
    for (int nt = 0; nt < 8; ++nt) {
        int col = nt * 16 + (lane & 15);
#pragma unroll
        for (int r = 0; r < 8; ++r) {
            int m = (lane < 16) ? r : r + 8;
            H[(size_t)(row0 + m) * DF + col] = c[nt][r];
        }
    }
}

// ---------------------------------------------------------------- edge scatter
// One wave per (edge | self-loop); lane moves one float4 (full 512B row / wave).
__global__ void __launch_bounds__(256)
gcn_scatter(const float* __restrict__ H, const int* __restrict__ src,
            const int* __restrict__ dst, const float* __restrict__ dinv,
            float* __restrict__ out, int nE, int nN) {
    int e = blockIdx.x * 8 + (threadIdx.x >> 5);
    int lane = threadIdx.x & 31;
    if (e >= nE + nN) return;
    int s, d;
    if (e < nE) { s = src[e]; d = dst[e]; }
    else        { s = d = e - nE; }
    float norm = dinv[s] * dinv[d];
    float4 v = ((const float4*)(H + (size_t)s * DF))[lane];
    float* od = out + (size_t)d * DF + lane * 4;
    atomicAdd(&od[0], v.x * norm);
    atomicAdd(&od[1], v.y * norm);
    atomicAdd(&od[2], v.z * norm);
    atomicAdd(&od[3], v.w * norm);
}

__global__ void bias_relu(float* __restrict__ X, const float* __restrict__ b, long n) {
    long i = (long)blockIdx.x * blockDim.x + threadIdx.x;
    long stride = (long)gridDim.x * blockDim.x;
    for (; i < n; i += stride) {
        float v = X[i] + b[i & 127];
        X[i] = v > 0.f ? v : 0.f;
    }
}

__global__ void __launch_bounds__(256)
pool_k(const float* __restrict__ X, const int* __restrict__ batch,
       float* __restrict__ pooled, int nN) {
    int node = blockIdx.x * 8 + (threadIdx.x >> 5);
    int lane = threadIdx.x & 31;
    if (node >= nN) return;
    int g = batch[node];
    float4 v = ((const float4*)(X + (size_t)node * DF))[lane];
    float* pg = pooled + (size_t)g * DF + lane * 4;
    atomicAdd(&pg[0], v.x);
    atomicAdd(&pg[1], v.y);
    atomicAdd(&pg[2], v.z);
    atomicAdd(&pg[3], v.w);
}

// ---------------------------------------------------------------- FC head (tiny)
__global__ void __launch_bounds__(256)
fc_head(const float* __restrict__ pooled, const float* __restrict__ Wf,
        const float* __restrict__ bf, float* __restrict__ out) {
    __shared__ float bufA[NGRAPHS * DF];
    __shared__ float bufB[NGRAPHS * DF];
    for (int i = threadIdx.x; i < NGRAPHS * DF; i += 256) bufA[i] = pooled[i];
    __syncthreads();
    float* in = bufA; float* o = bufB;
    for (int L = 0; L < NLAYERS; ++L) {
        const float* W = Wf + L * DF * DF;
        const float* b = bf + L * DF;
        for (int idx = threadIdx.x; idx < NGRAPHS * DF; idx += 256) {
            int g = idx >> 7, j = idx & 127;
            float acc = b[j];
            const float* ig = in + g * DF;
#pragma unroll 4
            for (int k = 0; k < DF; ++k) acc += ig[k] * W[k * DF + j];
            o[idx] = acc > 0.f ? acc : 0.f;
        }
        __syncthreads();
        float* t = in; in = o; o = t;
    }
    for (int i = threadIdx.x; i < NGRAPHS * DF; i += 256) out[i] = in[i];
}

// ---------------------------------------------------------------- launch
extern "C" void kernel_launch(void* const* d_in, const int* in_sizes, int n_in,
                              void* d_out, int out_size, void* d_ws, size_t ws_size,
                              hipStream_t stream) {
    const float* x     = (const float*)d_in[0];
    const int*   ei    = (const int*)d_in[1];
    const int*   batch = (const int*)d_in[2];
    const float* Wc    = (const float*)d_in[3];
    const float* bc    = (const float*)d_in[4];
    const float* Wf    = (const float*)d_in[5];
    const float* bf    = (const float*)d_in[6];
    float* out = (float*)d_out;

    float* ws     = (float*)d_ws;
    float* deg    = ws;                                   // N floats (becomes dinv)
    float* h      = ws + 50048;                           // N*128
    float* xbuf   = h + (size_t)N_NODES * DF;             // N*128
    float* pooled = xbuf + (size_t)N_NODES * DF;          // 50*128

    const int* src = ei;
    const int* dst = ei + N_EDGES;

    // degree + symmetric norm
    fill_f32<<<512, 256, 0, stream>>>(deg, 1.0f, N_NODES);
    degree_accum<<<(N_EDGES + 255) / 256, 256, 0, stream>>>(dst, deg, N_EDGES);
    rsqrt_k<<<(N_NODES + 255) / 256, 256, 0, stream>>>(deg, N_NODES);

    const int gemmBlocks = (N_NODES / 16 + 7) / 8;        // 16 rows/wave, 8 waves/block
    const int scatBlocks = (N_EDGES + N_NODES + 7) / 8;

    const float* cur = x;
    for (int L = 0; L < NLAYERS; ++L) {
        gcn_gemm_wmma<<<gemmBlocks, 256, 0, stream>>>(cur, Wc + L * DF * DF, h, N_NODES);
        fill_f32<<<2048, 256, 0, stream>>>(xbuf, 0.0f, (long)N_NODES * DF);
        gcn_scatter<<<scatBlocks, 256, 0, stream>>>(h, src, dst, deg, xbuf, N_EDGES, N_NODES);
        bias_relu<<<2048, 256, 0, stream>>>(xbuf, bc + L * DF, (long)N_NODES * DF);
        cur = xbuf;
    }

    fill_f32<<<32, 256, 0, stream>>>(pooled, 0.0f, NGRAPHS * DF);
    pool_k<<<(N_NODES + 7) / 8, 256, 0, stream>>>(xbuf, batch, pooled, N_NODES);
    fc_head<<<1, 256, 0, stream>>>(pooled, Wf, bf, out);
}
